// RNA_Model_66821101191602
// MI455X (gfx1250) — matrix operand order repaired and checked by reading, not compiled, not run on hardware
//
#include <hip/hip_runtime.h>
#include <hip/hip_bf16.h>

#define D_     192
#define DEPTH_ 12
#define H_     6
#define DH_    32
#define BINS_  65
#define B_     32
#define L_     512
#define BL_    (B_*L_)   /* 16384 rows */

typedef _Float16 h16;
typedef __attribute__((ext_vector_type(8)))  _Float16 v8h;
typedef __attribute__((ext_vector_type(16))) _Float16 v16h;
typedef __attribute__((ext_vector_type(8)))  float    v8f;

// ---- async global->LDS copy (CDNA5 GLOBAL_LOAD_ASYNC_TO_LDS_B128, ASYNCcnt) ----
// Inline asm per CDNA5 ISA (VDST = LDS byte address VGPR,
// VADDR = 64-bit global address VGPR pair, SADDR=off). The builtin on this
// toolchain has a descriptor-style signature, so we bypass it.
static __device__ __forceinline__ void async_cp16(const void* g, void* l) {
#if defined(__gfx1250__)
  unsigned lds_addr = (unsigned)(size_t)l;   // low 32 bits of flat addr = LDS offset
  asm volatile("global_load_async_to_lds_b128 %0, %1, off"
               :: "v"(lds_addr), "v"(g) : "memory");
#else
  *(v8h*)l = *(const v8h*)g;
#endif
}
static __device__ __forceinline__ void async_wait() {
#if defined(__gfx1250__)
  asm volatile("s_wait_asynccnt 0x0" ::: "memory");
#endif
}

// ---- WMMA fragment loaders (CDNA5 16-bit layouts, ISA 7.12.2) ----
// A (16x32 MxK): lane holds row m=lane%16; K chunks {c0..c0+7} and {c0+16..c0+23}, c0=(lane/16)*8
static __device__ __forceinline__ v16h load_frag_a(const h16* rowp, int lane) {
  const int c0 = (lane >> 4) << 3;
  v8h lo = *(const v8h*)(rowp + c0);
  v8h hi = *(const v8h*)(rowp + c0 + 16);
  return __builtin_shufflevector(lo, hi, 0,1,2,3,4,5,6,7,8,9,10,11,12,13,14,15);
}
// B (32x16 KxN): lane holds column n=lane%16; K chunks {c0..c0+7} and {c0+16..c0+23}, c0=(lane/16)*8
static __device__ __forceinline__ v16h load_frag_b(const h16* colp, int lane) {
  const int c0 = (lane >> 4) << 3;
  v8h lo = *(const v8h*)(colp + c0);
  v8h hi = *(const v8h*)(colp + c0 + 16);
  return __builtin_shufflevector(lo, hi, 0,1,2,3,4,5,6,7,8,9,10,11,12,13,14,15);
}

// ---- utility kernels -------------------------------------------------------
__global__ void cvt_f32_f16_kernel(const float* __restrict__ src, h16* __restrict__ dst, int n) {
  int i = blockIdx.x * 256 + threadIdx.x;
  if (i < n) dst[i] = (h16)src[i];
}

__global__ void embed_kernel(const int* __restrict__ seq, const float* __restrict__ emb,
                             float* __restrict__ x) {
  int i  = blockIdx.x * 256 + threadIdx.x;     // i < BL_*D_
  int bl = i / D_, d = i - bl * D_;
  x[i] = emb[seq[bl] * D_ + d];
}

// LayerNorm: one wave per row of 192, 8 rows per block; f16 output for WMMA.
__global__ void ln_kernel(const float* __restrict__ x, const float* __restrict__ s,
                          const float* __restrict__ b, h16* __restrict__ out) {
  int w = threadIdx.x >> 5, lane = threadIdx.x & 31;
  int row = blockIdx.x * 8 + w;
  const float* xr = x + (size_t)row * D_;
  float v[6]; float sum = 0.f, sq = 0.f;
  #pragma unroll
  for (int j = 0; j < 6; ++j) { v[j] = xr[lane + 32*j]; sum += v[j]; sq += v[j]*v[j]; }
  #pragma unroll
  for (int o = 16; o; o >>= 1) { sum += __shfl_xor(sum, o, 32); sq += __shfl_xor(sq, o, 32); }
  float mu  = sum * (1.0f / D_);
  float inv = rsqrtf(sq * (1.0f / D_) - mu*mu + 1e-5f);
  #pragma unroll
  for (int j = 0; j < 6; ++j) {
    int d = lane + 32*j;
    out[(size_t)row * D_ + d] = (h16)((v[j] - mu) * inv * s[d] + b[d]);
  }
}

// ---- generic WMMA GEMM: C[M,N] = A[M,K] @ W[N,K]^T + bias ------------------
// 8 waves/block; wave w computes rows m0..m0+15, block covers 128 x 64.
// All fragments for a k-step are loaded before the 4 WMMAs so loadcnt waits stage.
__global__ __launch_bounds__(256)
void gemm_wmma_kernel(const h16* __restrict__ A, const h16* __restrict__ W,
                      const float* __restrict__ bias,
                      h16* __restrict__ outH, float* __restrict__ outR,
                      int N, int K, int act) {
  int w = threadIdx.x >> 5, lane = threadIdx.x & 31;
  int m0 = blockIdx.y * 128 + w * 16;
  int n0 = blockIdx.x * 64;
  const h16* aRow = A + (size_t)(m0 + (lane & 15)) * K;
  const h16* wRow = W + (size_t)(n0 + (lane & 15)) * K;
  v8f acc[4] = {};
  for (int k0 = 0; k0 < K; k0 += 32) {
    if (k0 + 32 < K) __builtin_prefetch(aRow + k0 + 32, 0, 3);
    v16h a  = load_frag_a(aRow + k0, lane);
    v16h b0 = load_frag_b(wRow + (size_t)( 0) * K + k0, lane);
    v16h b1 = load_frag_b(wRow + (size_t)(16) * K + k0, lane);
    v16h b2 = load_frag_b(wRow + (size_t)(32) * K + k0, lane);
    v16h b3 = load_frag_b(wRow + (size_t)(48) * K + k0, lane);
    acc[0] = __builtin_amdgcn_wmma_f32_16x16x32_f16(false, a, false, b0, (short)0, acc[0], false, false);
    acc[1] = __builtin_amdgcn_wmma_f32_16x16x32_f16(false, a, false, b1, (short)0, acc[1], false, false);
    acc[2] = __builtin_amdgcn_wmma_f32_16x16x32_f16(false, a, false, b2, (short)0, acc[2], false, false);
    acc[3] = __builtin_amdgcn_wmma_f32_16x16x32_f16(false, a, false, b3, (short)0, acc[3], false, false);
  }
  int rbase = m0 + 8 * (lane >> 4);
  int cl = lane & 15;
  #pragma unroll
  for (int t = 0; t < 4; ++t) {
    int col = n0 + 16 * t + cl;
    float bv = bias[col];
    #pragma unroll
    for (int j = 0; j < 8; ++j) {
      size_t idx = (size_t)(rbase + j) * N + col;
      float v = acc[t][j] + bv;
      if (act) v = 0.5f * v * (1.0f + erff(v * 0.70710678118654752f));
      if (outH) outH[idx] = (h16)v;
      else      outR[idx] += v;
    }
  }
}

// ---- fused attention: scores(WMMA) + bias + mask + softmax + P@V(WMMA) -----
// grid: (L/64, H, B); block 256 (8 waves). Dynamic LDS: 260 KB of the 320 KB WGP LDS.
__global__ __launch_bounds__(256)
void attn_kernel(const h16* __restrict__ qkv, const float* __restrict__ bppm,
                 const unsigned char* __restrict__ mask,
                 const float* __restrict__ pair_w, const float* __restrict__ pair_b,
                 const float* __restrict__ relpos_w, const float* __restrict__ relpos_b,
                 h16* __restrict__ o16) {
  extern __shared__ char smem[];
  float* Sf = (float*)smem;                                  // 64 x 512 f32   (128 KB)
  h16*   P  = (h16*)(smem + 131072);                         // 64 x 512 f16   ( 64 KB)
  h16*   Ks = (h16*)(smem + 131072 + 65536);                 // 512 x 32 f16   ( 32 KB)
  h16*   Vt = (h16*)(smem + 131072 + 65536 + 32768);         // 32 x 512 f16   ( 32 KB)
  h16*   Qs = (h16*)(smem + 131072 + 65536 + 65536);         // 64 x 32  f16   (  4 KB)

  int q0 = blockIdx.x * 64, h = blockIdx.y, b = blockIdx.z;
  int tid = threadIdx.x, w = tid >> 5, lane = tid & 31;
  const h16* base = qkv + (size_t)b * L_ * 576;

  // stage Q via async DMA: 64 rows x 64 B -> 256 x 16 B chunks (1 per thread)
  {
    int r = tid >> 2, part = tid & 3;                     // row, 16B chunk
    async_cp16(base + (size_t)(q0 + r) * 576 + h * 32 + part * 8, Qs + r * 32 + part * 8);
  }
  // stage K via async DMA: 512 rows x 64 B -> 2048 x 16 B chunks (8 per thread)
  for (int c = tid; c < 512 * 4; c += 256) {
    int key = c >> 2, part = c & 3;
    async_cp16(base + (size_t)key * 576 + 192 + h * 32 + part * 8, Ks + key * 32 + part * 8);
  }
  // stage V transposed (d-major) -- elementwise transpose, coalesced global reads
  for (int i = tid; i < 512 * 32; i += 256) {
    int key = i >> 5, d = i & 31;
    Vt[d * 512 + key] = base[(size_t)key * 576 + 384 + h * 32 + d];
  }
  async_wait();
  __syncthreads();

  // phase 1: S = Q @ K^T via WMMA (qs = wave's q-subtile, kh = k-parity)
  int qs = w & 3, kh = w >> 2;
  v16h afr = load_frag_a(Qs + (size_t)(16 * qs + (lane & 15)) * 32, lane);
  for (int kt = 0; kt < 16; ++kt) {
    int ks = 2 * kt + kh;
    v16h bfr = load_frag_b(Ks + (size_t)(16 * ks + (lane & 15)) * 32, lane);
    v8f c = {};
    c = __builtin_amdgcn_wmma_f32_16x16x32_f16(false, afr, false, bfr, (short)0, c, false, false);
    int rb = 16 * qs + 8 * (lane >> 4), cb = 16 * ks + (lane & 15);
    #pragma unroll
    for (int j = 0; j < 8; ++j) Sf[(size_t)(rb + j) * 512 + cb] = c[j];
  }
  __syncthreads();

  // phase 2: scale + pair bias + relpos bias + mask (computed on the fly)
  const float scale = 0.17677669529663687f;  // 1/sqrt(32)
  float pw = pair_w[h], pb = pair_b[h], rb2 = relpos_b[h];
  const float* rw = relpos_w + h * BINS_;
  const float* bpp = bppm + ((size_t)b * L_ + q0) * L_;
  const unsigned char* mk = mask + (size_t)b * L_;
  for (int i = tid; i < 64 * 512; i += 256) {
    int r = i >> 9, col = i & 511;
    int rp = (q0 + r) - col; rp = rp < -32 ? -32 : (rp > 32 ? 32 : rp);
    float v = Sf[i] * scale + bpp[(size_t)r * L_ + col] * pw + pb + rw[rp + 32] + rb2;
    Sf[i] = mk[col] ? v : -1.0e9f;
  }
  __syncthreads();

  // phase 3: softmax, wave w owns rows 8w..8w+7; write f16 probs to P
  for (int rr = 0; rr < 8; ++rr) {
    int r = w * 8 + rr;
    float* Sr = Sf + (size_t)r * 512;
    float ev[16]; float m = -3.0e38f;
    #pragma unroll
    for (int j = 0; j < 16; ++j) { ev[j] = Sr[lane + 32*j]; m = fmaxf(m, ev[j]); }
    #pragma unroll
    for (int o = 16; o; o >>= 1) m = fmaxf(m, __shfl_xor(m, o, 32));
    float s = 0.f;
    #pragma unroll
    for (int j = 0; j < 16; ++j) { ev[j] = __expf(ev[j] - m); s += ev[j]; }
    #pragma unroll
    for (int o = 16; o; o >>= 1) s += __shfl_xor(s, o, 32);
    float inv = 1.0f / s;
    h16* Pr = P + (size_t)r * 512;
    #pragma unroll
    for (int j = 0; j < 16; ++j) Pr[lane + 32*j] = (h16)(ev[j] * inv);
  }
  __syncthreads();

  // phase 4: O = P @ V via WMMA; wave -> (q-subtile qs, d-half)
  int d0 = 16 * (w >> 2);
  v8f c = {};
  for (int kt = 0; kt < 16; ++kt) {
    v16h ap = load_frag_a(P  + (size_t)(16 * qs + (lane & 15)) * 512 + kt * 32, lane);
    v16h bv = load_frag_b(Vt + (size_t)(d0 + (lane & 15)) * 512 + kt * 32, lane);
    c = __builtin_amdgcn_wmma_f32_16x16x32_f16(false, ap, false, bv, (short)0, c, false, false);
  }
  int rq = q0 + 16 * qs + 8 * (lane >> 4);
  int dd = h * 32 + d0 + (lane & 15);
  #pragma unroll
  for (int j = 0; j < 8; ++j)
    o16[(size_t)(b * L_ + rq + j) * D_ + dd] = (h16)c[j];
}

// ---- final projection to 2 channels ---------------------------------------
__global__ void proj_kernel(const float* __restrict__ x, const float* __restrict__ pw,
                            const float* __restrict__ pb, float* __restrict__ out) {
  int w = threadIdx.x >> 5, lane = threadIdx.x & 31;
  int row = blockIdx.x * 8 + w;
  const float* xr = x + (size_t)row * D_;
  float s0 = 0.f, s1 = 0.f;
  #pragma unroll
  for (int j = 0; j < 6; ++j) {
    int d = lane + 32*j; float v = xr[d];
    s0 += v * pw[d]; s1 += v * pw[D_ + d];
  }
  #pragma unroll
  for (int o = 16; o; o >>= 1) { s0 += __shfl_xor(s0, o, 32); s1 += __shfl_xor(s1, o, 32); }
  if (lane == 0) { out[row*2 + 0] = s0 + pb[0]; out[row*2 + 1] = s1 + pb[1]; }
}

extern "C" void kernel_launch(void* const* d_in, const int* in_sizes, int n_in,
                              void* d_out, int out_size, void* d_ws, size_t ws_size,
                              hipStream_t stream) {
  (void)in_sizes; (void)n_in; (void)out_size; (void)ws_size;
  const int*           seq      = (const int*)d_in[0];
  const unsigned char* mask     = (const unsigned char*)d_in[1];
  const float*         bppm     = (const float*)d_in[2];
  const float*         emb      = (const float*)d_in[3];
  const float*         pair_w   = (const float*)d_in[4];
  const float*         pair_b   = (const float*)d_in[5];
  const float*         relpos_w = (const float*)d_in[6];
  const float*         relpos_b = (const float*)d_in[7];
  const float*         ln1_s    = (const float*)d_in[8];
  const float*         ln1_b    = (const float*)d_in[9];
  const float*         qkv_w    = (const float*)d_in[10];
  const float*         qkv_b    = (const float*)d_in[11];
  const float*         out_w    = (const float*)d_in[12];
  const float*         out_b    = (const float*)d_in[13];
  const float*         ln2_s    = (const float*)d_in[14];
  const float*         ln2_b    = (const float*)d_in[15];
  const float*         ff1_w    = (const float*)d_in[16];
  const float*         ff1_b    = (const float*)d_in[17];
  const float*         ff2_w    = (const float*)d_in[18];
  const float*         ff2_b    = (const float*)d_in[19];
  const float*         proj_w   = (const float*)d_in[20];
  const float*         proj_b   = (const float*)d_in[21];

  char* ws = (char*)d_ws;
  size_t off = 0;
  auto carve = [&](size_t bytes) { void* p = ws + off; off += (bytes + 255) & ~(size_t)255; return p; };
  float* x     = (float*)carve((size_t)BL_ * D_ * 4);
  h16*   hbuf  = (h16*)  carve((size_t)BL_ * D_ * 2);
  h16*   qkv16 = (h16*)  carve((size_t)BL_ * 576 * 2);
  h16*   o16   = (h16*)  carve((size_t)BL_ * D_ * 2);
  h16*   ff16  = (h16*)  carve((size_t)BL_ * 768 * 2);
  h16*   wqkv  = (h16*)  carve((size_t)DEPTH_ * 576 * D_ * 2);
  h16*   wout  = (h16*)  carve((size_t)DEPTH_ * D_ * D_ * 2);
  h16*   wff1  = (h16*)  carve((size_t)DEPTH_ * 768 * D_ * 2);
  h16*   wff2  = (h16*)  carve((size_t)DEPTH_ * D_ * 768 * 2);

  // weight f32 -> f16 conversion
  int n1 = DEPTH_ * 576 * D_, n2 = DEPTH_ * D_ * D_, n3 = DEPTH_ * 768 * D_;
  cvt_f32_f16_kernel<<<(n1 + 255) / 256, 256, 0, stream>>>(qkv_w, wqkv, n1);
  cvt_f32_f16_kernel<<<(n2 + 255) / 256, 256, 0, stream>>>(out_w, wout, n2);
  cvt_f32_f16_kernel<<<(n3 + 255) / 256, 256, 0, stream>>>(ff1_w, wff1, n3);
  cvt_f32_f16_kernel<<<(n3 + 255) / 256, 256, 0, stream>>>(ff2_w, wff2, n3);

  embed_kernel<<<(BL_ * D_) / 256, 256, 0, stream>>>(seq, emb, x);

  const size_t attn_lds = 131072 + 65536 + 32768 + 32768 + 4096;  // 260 KB
  for (int i = 0; i < DEPTH_; ++i) {
    ln_kernel<<<BL_ / 8, 256, 0, stream>>>(x, ln1_s + i * D_, ln1_b + i * D_, hbuf);
    gemm_wmma_kernel<<<dim3(576 / 64, BL_ / 128), 256, 0, stream>>>(
        hbuf, wqkv + (size_t)i * 576 * D_, qkv_b + i * 576, qkv16, nullptr, 576, D_, 0);
    attn_kernel<<<dim3(L_ / 64, H_, B_), 256, attn_lds, stream>>>(
        qkv16, bppm, mask, pair_w, pair_b, relpos_w, relpos_b, o16);
    gemm_wmma_kernel<<<dim3(D_ / 64, BL_ / 128), 256, 0, stream>>>(
        o16, wout + (size_t)i * D_ * D_, out_b + i * D_, nullptr, x, D_, D_, 0);
    ln_kernel<<<BL_ / 8, 256, 0, stream>>>(x, ln2_s + i * D_, ln2_b + i * D_, hbuf);
    gemm_wmma_kernel<<<dim3(768 / 64, BL_ / 128), 256, 0, stream>>>(
        hbuf, wff1 + (size_t)i * 768 * D_, ff1_b + i * 768, ff16, nullptr, 768, D_, 1);
    gemm_wmma_kernel<<<dim3(D_ / 64, BL_ / 128), 256, 0, stream>>>(
        ff16, wff2 + (size_t)i * D_ * 768, ff2_b + i * D_, nullptr, x, D_, 768, 0);
  }
  proj_kernel<<<BL_ / 8, 256, 0, stream>>>(x, proj_w, proj_b, (float*)d_out);
}
